// DiffusionPolicy_80144089743553
// MI455X (gfx1250) — matrix operand order repaired and checked by reading, not compiled, not run on hardware
//
#include <hip/hip_runtime.h>
#include <stdint.h>

// ---------------------------------------------------------------------------
// Diffusion policy sampler for MI455X (gfx1250, wave32, WMMA).
//
// Roofline: 2.1 TFLOP of GEMM vs ~25MB of HBM traffic -> pure matrix compute.
// Weights (312KB f16) live in LDS per workgroup (staged by the Tensor Data
// Mover); each wave owns 16 batch rows and keeps ALL activations in VGPRs
// across the 100-step loop.  GEMMs are done transposed (W^T as WMMA A-operand,
// activations^T as B) so the D->B relayout between layers is in-lane
// cvt_pkrtz + one half-wave shuffle.  A-fragment stream is software-pipelined
// one fragment ahead (partial s_wait_dscnt before each WMMA).
// ---------------------------------------------------------------------------

typedef _Float16 half_t;
typedef __attribute__((ext_vector_type(16))) _Float16 v16h;
typedef __attribute__((ext_vector_type(8)))  float    v8f;
typedef __attribute__((ext_vector_type(8)))  unsigned v8u;
typedef __attribute__((ext_vector_type(4)))  unsigned v4u;
typedef __attribute__((ext_vector_type(8)))  int      v8i;
typedef __attribute__((ext_vector_type(4)))  int      v4i;

#define OBS_DIM   64
#define ACT_DIM   16
#define HIDDEN    256
#define TIME_DIM  16
#define T_STEPS   100

// A-fragment storage: each frag = 16x32 f16 = 512 halves (32 lanes * 16).
#define W1_BASE   0      // KC=3,  NT=16 -> 48 frags
#define W2_BASE   48     // KC=8,  NT=16 -> 128 frags
#define W3_BASE   176    // KC=8,  NT=16 -> 128 frags
#define WF_BASE   304    // KC=8,  NT=1  -> 8 frags
#define TOT_FRAGS 312
#define W_HALVES  (TOT_FRAGS * 512)
#define W_BYTES   (W_HALVES * 2)          // 319488 bytes of LDS weights
#define W_QWORDS  (W_BYTES / 8)           // 39936 8-byte elements (fits 16-bit tile_dim0)

// ---------------------------------------------------------------------------
// helpers
// ---------------------------------------------------------------------------
__device__ __forceinline__ float mish_f(float x) {
  // mish(x) = x*tanh(softplus(x)) = x*(u^2-1)/(u^2+1), u = 1+exp(x)
  float e = __expf(x);
  float u = 1.0f + e;
  float s = u * u;
  float t = __builtin_amdgcn_rcpf(s + 1.0f);
  return x - 2.0f * x * t;   // x*(1 - 2/(s+1))
}

__device__ __forceinline__ unsigned pcg_hash(unsigned v) {
  v = v * 747796405u + 2891336453u;
  unsigned w = ((v >> ((v >> 28u) + 4u)) ^ v) * 277803737u;
  return (w >> 22u) ^ w;
}

__device__ __forceinline__ float hashnorm(unsigned a, unsigned b, unsigned c) {
  unsigned s1 = pcg_hash(a + 0x9E3779B9u * b + 0x85EBCA6Bu * c);
  unsigned s2 = pcg_hash(s1 ^ 0xB5297A4Du);
  float u1 = (float)(s1 >> 8) * (1.0f / 16777216.0f) + 1.0e-12f;
  float u2 = (float)(s2 >> 8) * (1.0f / 16777216.0f);
  float r  = __builtin_sqrtf(-2.0f * __logf(u1));
  return r * __cosf(6.28318530718f * u2);
}

// ---------------------------------------------------------------------------
// prep kernel 1: diffusion schedule (serial cumprod) + time embedding MLP for
// all 100 timesteps (batch-independent -> compute once).
// ---------------------------------------------------------------------------
__global__ void prep_sched_te(const float* __restrict__ tw1, const float* __restrict__ tb1,
                              const float* __restrict__ tw2, const float* __restrict__ tb2,
                              float* __restrict__ sched, float* __restrict__ teAll) {
  int tid = threadIdx.x;
  if (tid < T_STEPS) {
    int t = tid;
    float e[TIME_DIM];
    const float neglog = -__logf(10000.0f) / 7.0f;  // TIME_DIM/2 - 1 = 7
    #pragma unroll
    for (int h = 0; h < 8; ++h) {
      float f = __expf((float)h * neglog);
      float a = (float)t * f;
      e[h]     = __sinf(a);
      e[8 + h] = __cosf(a);
    }
    float hid[2 * TIME_DIM];
    for (int j = 0; j < 32; ++j) {
      float s = tb1[j];
      for (int k = 0; k < 16; ++k) s += e[k] * tw1[k * 32 + j];
      hid[j] = mish_f(s);
    }
    for (int o = 0; o < 16; ++o) {
      float s = tb2[o];
      for (int j = 0; j < 32; ++j) s += hid[j] * tw2[j * 16 + o];
      teAll[t * 16 + o] = s;
    }
  }
  if (tid == T_STEPS) {
    float ac = 1.0f;
    for (int i = 0; i < T_STEPS; ++i) {
      float beta  = 1e-4f + (0.02f - 1e-4f) * (float)i / (float)(T_STEPS - 1);
      float alpha = 1.0f - beta;
      float acp   = ac;            // cumprod up to i-1 (ac_prev[0] = 1)
      ac *= alpha;
      sched[i]         = sqrtf(1.0f / ac);
      sched[100 + i]   = sqrtf((1.0f - ac) / ac);
      sched[200 + i]   = beta * sqrtf(acp) / (1.0f - ac);
      sched[300 + i]   = (1.0f - acp) * sqrtf(alpha) / (1.0f - ac);
      float pv = beta * (1.0f - acp) / (1.0f - ac);
      sched[400 + i]   = __logf(fmaxf(pv, 1e-20f));
    }
  }
}

// ---------------------------------------------------------------------------
// prep kernel 2: transpose + f16-convert all weights into a
// 16-bit A-fragment layout, contiguous K (lanes 0-15: M, halves K 0-15; lanes 16-31:
// halves K 16-31) so the main kernel loads them with contiguous b128s.
// ---------------------------------------------------------------------------
__global__ void prep_weights(const float* __restrict__ w1, const float* __restrict__ w2,
                             const float* __restrict__ w3, const float* __restrict__ wf,
                             half_t* __restrict__ dst) {
  int g = blockIdx.x * blockDim.x + threadIdx.x;
  if (g >= TOT_FRAGS * 512) return;
  int frag = g >> 9, r = g & 511, lane = r >> 4, p = r & 15;
  const float* src; int KC, Nout, base;
  if (frag < W2_BASE)      { src = w1; KC = 3; Nout = 256; base = W1_BASE; }
  else if (frag < W3_BASE) { src = w2; KC = 8; Nout = 256; base = W2_BASE; }
  else if (frag < WF_BASE) { src = w3; KC = 8; Nout = 256; base = W3_BASE; }
  else                     { src = wf; KC = 8; Nout = 16;  base = WF_BASE; }
  int local = frag - base;
  int nt = local / KC, kc = local % KC;
  int k = 32 * kc + p + ((lane & 16) ? 16 : 0);                      // A-layout K
  int o = nt * 16 + (lane & 15);                                    // A-layout M
  dst[g] = (half_t)src[k * Nout + o];    // A = W^T : element (o,k) = W[k][o]
}

// ---------------------------------------------------------------------------
// one MLP layer: D(nt) = W^T_tile x B + bias, mish, repack into next-layer
// B fragments.  Output tiles are processed in even/odd pairs: one B fragment
// (K=32) of the next layer spans two 16-feature D tiles; halves owned by the
// partner half-wave arrive via __shfl_xor(.,16).
// ---------------------------------------------------------------------------
template<int KC>
__device__ __forceinline__ void mlp_layer(const half_t* __restrict__ sw,
                                          const float* __restrict__ bias,
                                          const unsigned (&bin)[KC][8],
                                          unsigned (&hout)[8][8],
                                          int lane, bool low) {
  const half_t* base = sw + lane * 16;
  v16h aCur = *(const v16h*)base;                  // fragment 0
  #pragma unroll
  for (int q = 0; q < 8; ++q) {
    unsigned evenPk[4], evenRv[4], oddPk[4], oddRv[4];
    #pragma unroll
    for (int e = 0; e < 2; ++e) {
      const int nt = 2 * q + e;
      v8f c;
      const float* bp = bias + nt * 16 + (low ? 0 : 8);
      #pragma unroll
      for (int r = 0; r < 8; ++r) c[r] = bp[r];
      #pragma unroll
      for (int kc = 0; kc < KC; ++kc) {
        const int fi = nt * KC + kc;
        v16h aNext = aCur;
        if (fi + 1 < 16 * KC)                       // prefetch next fragment
          aNext = *(const v16h*)(base + (fi + 1) * 512);
        v8u bv;
        #pragma unroll
        for (int i = 0; i < 8; ++i) bv[i] = bin[kc][i];
        v16h b = __builtin_bit_cast(v16h, bv);
        c = __builtin_amdgcn_wmma_f32_16x16x32_f16(false, aCur, false, b,
                                                   (short)0, c, false, false);
        aCur = aNext;
      }
      #pragma unroll
      for (int i = 0; i < 4; ++i) {
        float m0 = mish_f(c[2 * i]);
        float m1 = mish_f(c[2 * i + 1]);
        unsigned pk = __builtin_bit_cast(unsigned, __builtin_amdgcn_cvt_pkrtz(m0, m1));
        unsigned rv = (unsigned)__shfl_xor((int)pk, 16, 32);
        if (e == 0) { evenPk[i] = pk; evenRv[i] = rv; }
        else        { oddPk[i]  = pk; oddRv[i]  = rv; }
      }
    }
    #pragma unroll
    for (int i = 0; i < 4; ++i) {   // B layout: lanes<16 hold K 0-15, lanes>=16 K 16-31
      hout[q][i]     = low ? evenPk[i] : oddRv[i];
      hout[q][4 + i] = low ? evenRv[i] : oddPk[i];
    }
  }
}

// ---------------------------------------------------------------------------
// main kernel: 4 waves/WG, 16 batch rows per wave, 100 steps in registers.
// ---------------------------------------------------------------------------
__global__ __launch_bounds__(128, 1)
void diffusion_main(const float* __restrict__ state,
                    const half_t* __restrict__ wsW,
                    const float* __restrict__ b1, const float* __restrict__ b2,
                    const float* __restrict__ b3, const float* __restrict__ bf,
                    const float* __restrict__ sched, const float* __restrict__ teAll,
                    float* __restrict__ out) {
  extern __shared__ __align__(16) char smem[];
  half_t* sW = (half_t*)smem;
  float*  sB = (float*)(smem + W_BYTES);   // [0]=b1, [256]=b2, [512]=b3, [768]=bf

#if __has_builtin(__builtin_amdgcn_tensor_load_to_lds)
  // --- Tensor Data Mover: DMA the whole 312KB weight block into LDS with a
  // single descriptor (1 row x 39936 8-byte elements), issued by wave 0 only
  // (TDM ignores EXEC; waves 1-3 branch around it).  Tracked by TENSORcnt.
  if (threadIdx.x < 32) {
    unsigned long long ga = (unsigned long long)(const void*)wsW;
    v4u g0;
    g0[0] = 1u;                                          // count=1, user D#
    g0[1] = 0u;                                          // lds_addr = 0
    g0[2] = (unsigned)(ga & 0xFFFFFFFFu);                // global_addr[31:0]
    g0[3] = (unsigned)((ga >> 32) & 0x01FFFFFFu) | (2u << 30);  // addr[56:32] | type=2
    v8i g1;
    g1[0] = (int)(3u << 16);                             // data_size = 3 (8B), no mask
    g1[1] = (int)((W_QWORDS & 0xFFFFu) << 16);           // tensor_dim0[15:0]
    g1[2] = (int)((W_QWORDS >> 16) | (1u << 16));        // tensor_dim0[31:16] | tensor_dim1=1
    g1[3] = (int)((W_QWORDS & 0xFFFFu) << 16);           // tile_dim0 = 39936
    g1[4] = 0;                                           // tile_dim1/2 unused
    g1[5] = (int)W_QWORDS;                               // tensor_dim0_stride
    g1[6] = 0; g1[7] = 0;
    v4i gz4 = {0, 0, 0, 0};                              // groups 2/3 unused (<=2D)
    v8i gz8 = {0, 0, 0, 0, 0, 0, 0, 0};                  // group 4 (6-arg toolchain form)
    __builtin_amdgcn_tensor_load_to_lds(g0, g1, gz4, gz4, gz8, 0);
    __builtin_amdgcn_s_wait_tensorcnt(0);
  }
#else
  {  // fallback: cooperative vector copy
    const uint4* src = (const uint4*)wsW;
    uint4* dst = (uint4*)smem;
    for (int i = threadIdx.x; i < W_BYTES / 16; i += 128) dst[i] = src[i];
  }
#endif
  for (int i = threadIdx.x; i < 256; i += 128) {
    sB[i] = b1[i]; sB[256 + i] = b2[i]; sB[512 + i] = b3[i];
  }
  if (threadIdx.x < 16) sB[768 + threadIdx.x] = bf[threadIdx.x];
  __syncthreads();

  const int  lane = threadIdx.x & 31;
  const int  wave = threadIdx.x >> 5;
  const bool low  = lane < 16;
  const int  off  = low ? 0 : 8;                    // feature offset this half-wave owns
  const long n_g  = (long)blockIdx.x * 64 + wave * 16 + (lane & 15);

  // --- resident input B-fragments: [x(16) | state(64) | te(16)] over K=96 ---
  // lanes >= 16 of chunk0 hold state[0..15]  (never overwritten),
  // lanes <  16 of chunk2 hold state[48..63] (never overwritten): per-step
  // updates self-select against the old value, no shadow copies needed.
  unsigned binIn[3][8];
  {
    const float* st = state + n_g * (long)OBS_DIM;
    #pragma unroll
    for (int c = 0; c < 3; ++c) {
      #pragma unroll
      for (int w = 0; w < 8; ++w) {
        int d0 = 32 * c + 2 * w + (low ? 0 : 16);   // input dim of half 2w
        float f0 = (d0     >= 16 && d0     < 80) ? st[d0 - 16] : 0.0f;
        float f1 = (d0 + 1 >= 16 && d0 + 1 < 80) ? st[d0 - 15] : 0.0f;
        binIn[c][w] = __builtin_bit_cast(unsigned, __builtin_amdgcn_cvt_pkrtz(f0, f1));
      }
    }
  }

  // --- x init (pseudo-normal, deterministic) ---
  float xv[8];
  #pragma unroll
  for (int r = 0; r < 8; ++r)
    xv[r] = hashnorm(7u, (unsigned)n_g, (unsigned)(off + r));

  auto packX = [&]() {   // fold updated x back into binIn[0] (lanes<16 slots only)
    unsigned xp[4], xr[4];
    #pragma unroll
    for (int i = 0; i < 4; ++i) {
      xp[i] = __builtin_bit_cast(unsigned,
                __builtin_amdgcn_cvt_pkrtz(xv[2 * i], xv[2 * i + 1]));
      xr[i] = (unsigned)__shfl_xor((int)xp[i], 16, 32);
    }
    #pragma unroll
    for (int i = 0; i < 4; ++i) {
      binIn[0][i]     = low ? xp[i] : binIn[0][i];
      binIn[0][4 + i] = low ? xr[i] : binIn[0][4 + i];
    }
  };
  packX();

  unsigned hA[8][8], hB[8][8];

  for (int t = T_STEPS - 1; t >= 0; --t) {
    // --- inject time embedding (uniform loads, branchless select) ---
    #pragma unroll
    for (int w = 0; w < 8; ++w) {
      float t0 = teAll[t * 16 + 2 * w];
      float t1 = teAll[t * 16 + 2 * w + 1];
      unsigned tw = __builtin_bit_cast(unsigned, __builtin_amdgcn_cvt_pkrtz(t0, t1));
      binIn[2][w] = low ? binIn[2][w] : tw;         // hi-lane slots carry te
    }

    // --- 3 hidden layers + final projection, all WMMA ---
    mlp_layer<3>(sW + W1_BASE * 512, sB,        binIn, hA, lane, low);
    mlp_layer<8>(sW + W2_BASE * 512, sB + 256,  hA,    hB, lane, low);
    mlp_layer<8>(sW + W3_BASE * 512, sB + 512,  hB,    hA, lane, low);

    v8f c;
    {
      const half_t* base = sW + WF_BASE * 512 + lane * 16;
      v16h aCur = *(const v16h*)base;
      const float* bp = sB + 768 + off;
      #pragma unroll
      for (int r = 0; r < 8; ++r) c[r] = bp[r];
      #pragma unroll
      for (int kc = 0; kc < 8; ++kc) {
        v16h aNext = aCur;
        if (kc + 1 < 8) aNext = *(const v16h*)(base + (kc + 1) * 512);
        v8u bv;
        #pragma unroll
        for (int i = 0; i < 8; ++i) bv[i] = hA[kc][i];
        v16h b = __builtin_bit_cast(v16h, bv);
        c = __builtin_amdgcn_wmma_f32_16x16x32_f16(false, aCur, false, b,
                                                   (short)0, c, false, false);
        aCur = aNext;
      }
    }

    // --- DDPM posterior step (elementwise, in-lane) ---
    float s_sr   = sched[t];
    float s_srm1 = sched[100 + t];
    float s_c0   = sched[200 + t];
    float s_cxt  = sched[300 + t];
    float sig    = (t != 0) ? __expf(0.5f * sched[400 + t]) : 0.0f;
    #pragma unroll
    for (int r = 0; r < 8; ++r) {
      float eps = c[r], x = xv[r];
      float x0   = fminf(1.0f, fmaxf(-1.0f, s_sr * x - s_srm1 * eps));
      float mean = s_c0 * x0 + s_cxt * x;
      float z    = hashnorm(0x2Au ^ (0x9E3779B9u * (unsigned)t),
                            (unsigned)n_g, (unsigned)(off + r));
      xv[r] = mean + sig * z;
    }
    packX();
  }

  #pragma unroll
  for (int r = 0; r < 8; ++r)
    out[n_g * (long)ACT_DIM + off + r] = fminf(1.0f, fmaxf(-1.0f, xv[r]));
}

// ---------------------------------------------------------------------------
extern "C" void kernel_launch(void* const* d_in, const int* in_sizes, int n_in,
                              void* d_out, int out_size, void* d_ws, size_t ws_size,
                              hipStream_t stream) {
  const float* state   = (const float*)d_in[0];
  const float* time_w1 = (const float*)d_in[1];
  const float* time_b1 = (const float*)d_in[2];
  const float* time_w2 = (const float*)d_in[3];
  const float* time_b2 = (const float*)d_in[4];
  const float* mid_w1  = (const float*)d_in[5];
  const float* mid_b1  = (const float*)d_in[6];
  const float* mid_w2  = (const float*)d_in[7];
  const float* mid_b2  = (const float*)d_in[8];
  const float* mid_w3  = (const float*)d_in[9];
  const float* mid_b3  = (const float*)d_in[10];
  const float* final_w = (const float*)d_in[11];
  const float* final_b = (const float*)d_in[12];
  float* out = (float*)d_out;

  half_t* wsW     = (half_t*)d_ws;
  float*  wsSched = (float*)((char*)d_ws + W_BYTES);   // 500 floats
  float*  wsTe    = wsSched + 512;                     // 1600 floats

  int B = in_sizes[0] / OBS_DIM;   // 65536

  prep_sched_te<<<1, 128, 0, stream>>>(time_w1, time_b1, time_w2, time_b2,
                                       wsSched, wsTe);
  prep_weights<<<(TOT_FRAGS * 512 + 255) / 256, 256, 0, stream>>>(
      mid_w1, mid_w2, mid_w3, final_w, wsW);

  size_t ldsBytes = (size_t)W_BYTES + 784 * sizeof(float);  // weights + biases
  diffusion_main<<<B / 64, 128, ldsBytes, stream>>>(
      state, wsW, mid_b1, mid_b2, mid_b3, final_b, wsSched, wsTe, out);
}
